// Temporal_map_80951543595343
// MI455X (gfx1250) — matrix operand order repaired and checked by reading, not compiled, not run on hardware
//
#include <hip/hip_runtime.h>
#include <hip/hip_bf16.h>

// ---------------------------------------------------------------------------
// Problem constants (from reference): B=16, T=32, C=64, H=W=16, NUM_FEAT=10
//   M = B*T = 512, K = C*H*W = 16384, N = NUM_FEAT*H*W = 2560
// out[m,n] = gates[m] * ( x[m,:] . G3_w[n,:] + G3_b[n] ),  42.9 GFLOP GEMM
// ---------------------------------------------------------------------------
#define M_DIM 512
#define K_DIM 16384
#define N_DIM 2560

typedef __attribute__((ext_vector_type(16))) __bf16 v16bf;
typedef __attribute__((ext_vector_type(8)))  __bf16 v8bf;
typedef __attribute__((ext_vector_type(8)))  float  v8f;

// ---------------------------------------------------------------------------
// Kernel 1: wbar[c] = mean over 10 filters of conv1_w[f][c]
// ---------------------------------------------------------------------------
__global__ __launch_bounds__(256) void wbar_kernel(const float* __restrict__ c1w,
                                                   float* __restrict__ wbar) {
    int c = blockIdx.x * 256 + threadIdx.x;
    if (c < K_DIM) {
        float s = 0.f;
#pragma unroll
        for (int f = 0; f < 10; ++f) s += c1w[f * K_DIM + c];
        wbar[c] = s * 0.1f;
    }
}

// ---------------------------------------------------------------------------
// Kernel 2: pooled[m] = x[m,:] . wbar + mean(conv1_b)   (one block per row)
// ---------------------------------------------------------------------------
__global__ __launch_bounds__(256) void pooled_kernel(const float* __restrict__ x,
                                                     const float* __restrict__ wbar,
                                                     const float* __restrict__ c1b,
                                                     float* __restrict__ pooled) {
    const int m = blockIdx.x;
    const float* row = x + (size_t)m * K_DIM;
    float s = 0.f;
    for (int c = threadIdx.x * 4; c < K_DIM; c += 256 * 4) {
        float4 a = *(const float4*)(row + c);
        float4 w = *(const float4*)(wbar + c);
        s += a.x * w.x + a.y * w.y + a.z * w.z + a.w * w.w;
    }
    __shared__ float red[256];
    red[threadIdx.x] = s;
    __syncthreads();
#pragma unroll
    for (int off = 128; off > 0; off >>= 1) {
        if (threadIdx.x < off) red[threadIdx.x] += red[threadIdx.x + off];
        __syncthreads();
    }
    if (threadIdx.x == 0) {
        float bb = 0.f;
#pragma unroll
        for (int f = 0; f < 10; ++f) bb += c1b[f];
        pooled[m] = red[0] + bb * 0.1f;
    }
}

// ---------------------------------------------------------------------------
// Kernel 3: z = pooled @ ffnn1_w^T + b ; gates = a0*relu+a1*sigmoid+a2*softmax_T
// T == warpSize == 32 on gfx1250, so softmax over T is intra-wave shuffles.
// ---------------------------------------------------------------------------
__global__ __launch_bounds__(512) void gates_kernel(const float* __restrict__ pooled,
                                                    const float* __restrict__ f1w,
                                                    const float* __restrict__ f1b,
                                                    const float* __restrict__ aw,
                                                    float* __restrict__ gates) {
    __shared__ float ps[M_DIM];
    const int tid = threadIdx.x;
    ps[tid] = pooled[tid];
    __syncthreads();

    const int b = tid >> 5;
    const int s = tid & 31;
    const float* prow = ps + b * 32;
    const float* wrow = f1w + s * 32;     // ffnn1_w[s][t]
    float z = f1b[s];
#pragma unroll
    for (int t = 0; t < 32; ++t) z += prow[t] * wrow[t];

    float mx = z;
#pragma unroll
    for (int off = 16; off > 0; off >>= 1) mx = fmaxf(mx, __shfl_xor(mx, off, 32));
    float e = __expf(z - mx);
    float sum = e;
#pragma unroll
    for (int off = 16; off > 0; off >>= 1) sum += __shfl_xor(sum, off, 32);
    const float soft = e / sum;

    const float relu = z > 0.f ? z : 0.f;
    const float sig  = 1.f / (1.f + __expf(-z));
    gates[tid] = aw[0] * relu + aw[1] * sig + aw[2] * soft;
}

// ---------------------------------------------------------------------------
// Kernel 4a: bulk fp32 -> bf16 conversion (8 elems / thread, 16B stores)
// ---------------------------------------------------------------------------
__global__ __launch_bounds__(256) void cvt_bf16_kernel(const float* __restrict__ src,
                                                       __bf16* __restrict__ dst,
                                                       int n8) {
    int i = blockIdx.x * 256 + threadIdx.x;
    if (i < n8) {
        const float4* p = (const float4*)src + (size_t)i * 2;
        float4 a = p[0], b = p[1];
        v8bf r;
        r[0] = (__bf16)a.x; r[1] = (__bf16)a.y; r[2] = (__bf16)a.z; r[3] = (__bf16)a.w;
        r[4] = (__bf16)b.x; r[5] = (__bf16)b.y; r[6] = (__bf16)b.z; r[7] = (__bf16)b.w;
        *((v8bf*)dst + i) = r;
    }
}

// ---------------------------------------------------------------------------
// Fragment loaders in the documented gfx1250 WMMA VGPR layouts.
// A 16x32 bf16: lane&15 = M row; elements[0..7] = K half*8.., [8..15] = K 16+half*8..
// B 32x16 bf16: lane&15 = N col; elements[0..7] = K half*8.., [8..15] = K 16+half*8..
// (G3_w is [N][K] row-major, so a B column == contiguous weight row strip).
// ---------------------------------------------------------------------------
__device__ __forceinline__ v16bf load_frag_a_bf(const __bf16* __restrict__ row, int k0, int half) {
    v8bf lo = *(const v8bf*)(row + k0 + half * 8);
    v8bf hi = *(const v8bf*)(row + k0 + 16 + half * 8);
    return __builtin_shufflevector(lo, hi, 0, 1, 2, 3, 4, 5, 6, 7,
                                   8, 9, 10, 11, 12, 13, 14, 15);
}
__device__ __forceinline__ v16bf load_frag_b_bf(const __bf16* __restrict__ row, int k0, int half) {
    v8bf lo = *(const v8bf*)(row + k0 + half * 8);
    v8bf hi = *(const v8bf*)(row + k0 + 16 + half * 8);
    return __builtin_shufflevector(lo, hi, 0, 1, 2, 3, 4, 5, 6, 7,
                                   8, 9, 10, 11, 12, 13, 14, 15);
}

// fp32 variants (fallback path when workspace is too small to pre-convert)
__device__ __forceinline__ v16bf cvt16(float4 c0, float4 c1, float4 c2, float4 c3) {
    v16bf r;
    r[0]  = (__bf16)c0.x; r[1]  = (__bf16)c0.y; r[2]  = (__bf16)c0.z; r[3]  = (__bf16)c0.w;
    r[4]  = (__bf16)c1.x; r[5]  = (__bf16)c1.y; r[6]  = (__bf16)c1.z; r[7]  = (__bf16)c1.w;
    r[8]  = (__bf16)c2.x; r[9]  = (__bf16)c2.y; r[10] = (__bf16)c2.z; r[11] = (__bf16)c2.w;
    r[12] = (__bf16)c3.x; r[13] = (__bf16)c3.y; r[14] = (__bf16)c3.z; r[15] = (__bf16)c3.w;
    return r;
}
__device__ __forceinline__ v16bf load_frag_a_f32(const float* __restrict__ row, int k0, int half) {
    const float4* p = (const float4*)(row + k0 + half * 8);
    const float4* q = (const float4*)(row + k0 + 16 + half * 8);
    return cvt16(p[0], p[1], q[0], q[1]);
}
__device__ __forceinline__ v16bf load_frag_b_f32(const float* __restrict__ row, int k0, int half) {
    const float4* p = (const float4*)(row + k0 + half * 8);
    const float4* q = (const float4*)(row + k0 + 16 + half * 8);
    return cvt16(p[0], p[1], q[0], q[1]);
}

// ---------------------------------------------------------------------------
// Shared epilogue: C/D layout col = n0+ni*16+(lane&15), row = m0+mi*16+half*8+r
// ---------------------------------------------------------------------------
__device__ __forceinline__ void store_gated(v8f acc00, v8f acc01, v8f acc10, v8f acc11,
                                            const float* __restrict__ g3b,
                                            const float* __restrict__ gates,
                                            float* __restrict__ out,
                                            int m0, int n0, int half, int l16) {
    v8f accs[2][2];
    accs[0][0] = acc00; accs[0][1] = acc01; accs[1][0] = acc10; accs[1][1] = acc11;
#pragma unroll
    for (int mi = 0; mi < 2; ++mi) {
        const int rbase = m0 + mi * 16 + half * 8;
#pragma unroll
        for (int ni = 0; ni < 2; ++ni) {
            const int col = n0 + ni * 16 + l16;
            const float bias = g3b[col];
            float* op = out + (size_t)rbase * N_DIM + col;
#pragma unroll
            for (int r = 0; r < 8; ++r) {
                const float gv = gates[rbase + r];
                op[(size_t)r * N_DIM] = gv * (accs[mi][ni][r] + bias);
            }
        }
    }
}

// ---------------------------------------------------------------------------
// Kernel 4b: bf16 WMMA GEMM (preferred path). 4 waves/block, 64x64 block tile,
// 32x32 per-wave (2x2 v_wmma_f32_16x16x32_bf16 accumulators).
// Outer loop steps K by 128: one branchless clamped prefetch group, then
// 16 fragment loads into distinct registers, then 16 back-to-back WMMAs —
// wide window so loads overlap the matrix pipe instead of draining loadcnt.
// ---------------------------------------------------------------------------
__global__ __launch_bounds__(128) void gemm_bf16_kernel(const __bf16* __restrict__ xb,
                                                        const __bf16* __restrict__ gwb,
                                                        const float* __restrict__ g3b,
                                                        const float* __restrict__ gates,
                                                        float* __restrict__ out) {
    const int tid  = threadIdx.x;
    const int lane = tid & 31;
    const int wave = tid >> 5;
    const int half = lane >> 4;
    const int l16  = lane & 15;

    const int m0 = blockIdx.y * 64 + (wave >> 1) * 32;
    const int n0 = blockIdx.x * 64 + (wave & 1) * 32;

    const __bf16* arow0 = xb  + (size_t)(m0 + l16)      * K_DIM;
    const __bf16* arow1 = xb  + (size_t)(m0 + 16 + l16) * K_DIM;
    const __bf16* brow0 = gwb + (size_t)(n0 + l16)      * K_DIM;
    const __bf16* brow1 = gwb + (size_t)(n0 + 16 + l16) * K_DIM;

    v8f acc00 = {}, acc01 = {}, acc10 = {}, acc11 = {};

    for (int kb = 0; kb < K_DIM; kb += 128) {
        // branchless, always-in-range prefetch ~1KB ahead (global_prefetch_b8)
        const int pk = (kb + 512 < K_DIM) ? (kb + 512) : (K_DIM - 512);
        __builtin_prefetch(arow0 + pk, 0, 1);
        __builtin_prefetch(arow1 + pk, 0, 1);
        __builtin_prefetch(brow0 + pk, 0, 1);
        __builtin_prefetch(brow1 + pk, 0, 1);

        v16bf af0[4], af1[4], bf0[4], bf1[4];
#pragma unroll
        for (int u = 0; u < 4; ++u) {
            const int k0 = kb + u * 32;
            af0[u] = load_frag_a_bf(arow0, k0, half);
            af1[u] = load_frag_a_bf(arow1, k0, half);
            bf0[u] = load_frag_b_bf(brow0, k0, half);
            bf1[u] = load_frag_b_bf(brow1, k0, half);
        }
#pragma unroll
        for (int u = 0; u < 4; ++u) {
            acc00 = __builtin_amdgcn_wmma_f32_16x16x32_bf16(false, af0[u], false, bf0[u], (short)0, acc00, false, false);
            acc01 = __builtin_amdgcn_wmma_f32_16x16x32_bf16(false, af0[u], false, bf1[u], (short)0, acc01, false, false);
            acc10 = __builtin_amdgcn_wmma_f32_16x16x32_bf16(false, af1[u], false, bf0[u], (short)0, acc10, false, false);
            acc11 = __builtin_amdgcn_wmma_f32_16x16x32_bf16(false, af1[u], false, bf1[u], (short)0, acc11, false, false);
        }
    }
    store_gated(acc00, acc01, acc10, acc11, g3b, gates, out, m0, n0, half, l16);
}

// ---------------------------------------------------------------------------
// Kernel 4c: fused-convert GEMM fallback (used only if workspace is small)
// ---------------------------------------------------------------------------
__global__ __launch_bounds__(128) void gemm_fused_kernel(const float* __restrict__ x,
                                                         const float* __restrict__ g3w,
                                                         const float* __restrict__ g3b,
                                                         const float* __restrict__ gates,
                                                         float* __restrict__ out) {
    const int tid  = threadIdx.x;
    const int lane = tid & 31;
    const int wave = tid >> 5;
    const int half = lane >> 4;
    const int l16  = lane & 15;

    const int m0 = blockIdx.y * 64 + (wave >> 1) * 32;
    const int n0 = blockIdx.x * 64 + (wave & 1) * 32;

    const float* arow0 = x   + (size_t)(m0 + l16)      * K_DIM;
    const float* arow1 = x   + (size_t)(m0 + 16 + l16) * K_DIM;
    const float* brow0 = g3w + (size_t)(n0 + l16)      * K_DIM;
    const float* brow1 = g3w + (size_t)(n0 + 16 + l16) * K_DIM;

    v8f acc00 = {}, acc01 = {}, acc10 = {}, acc11 = {};

    for (int k0 = 0; k0 < K_DIM; k0 += 32) {
        v16bf a0 = load_frag_a_f32(arow0, k0, half);
        v16bf a1 = load_frag_a_f32(arow1, k0, half);
        v16bf b0 = load_frag_b_f32(brow0, k0, half);
        v16bf b1 = load_frag_b_f32(brow1, k0, half);

        acc00 = __builtin_amdgcn_wmma_f32_16x16x32_bf16(false, a0, false, b0, (short)0, acc00, false, false);
        acc01 = __builtin_amdgcn_wmma_f32_16x16x32_bf16(false, a0, false, b1, (short)0, acc01, false, false);
        acc10 = __builtin_amdgcn_wmma_f32_16x16x32_bf16(false, a1, false, b0, (short)0, acc10, false, false);
        acc11 = __builtin_amdgcn_wmma_f32_16x16x32_bf16(false, a1, false, b1, (short)0, acc11, false, false);
    }
    store_gated(acc00, acc01, acc10, acc11, g3b, gates, out, m0, n0, half, l16);
}

// ---------------------------------------------------------------------------
// Host launcher.  Inputs (setup_inputs order):
//  0 input [512,16384] f32  1 input1 (unused)  2 act_weights [3]
//  3 conv1_w [10,16384]  4 conv1_b [10]  5 G3_w [2560,16384]
//  6 G3_b [2560]  7 ffnn1_w [32,32]  8 ffnn1_b [32]
// Output: [512, 2560] f32 row-major.
// ws layout: [0, 69632)      : wbar(16384f) | pooled(512f) | gates(512f)
//            [69632, ...)    : x_bf16 (16 MB) | G3_w_bf16 (80 MB)   (if it fits)
// ---------------------------------------------------------------------------
extern "C" void kernel_launch(void* const* d_in, const int* in_sizes, int n_in,
                              void* d_out, int out_size, void* d_ws, size_t ws_size,
                              hipStream_t stream) {
    const float* input = (const float*)d_in[0];
    const float* aw    = (const float*)d_in[2];
    const float* c1w   = (const float*)d_in[3];
    const float* c1b   = (const float*)d_in[4];
    const float* g3w   = (const float*)d_in[5];
    const float* g3b   = (const float*)d_in[6];
    const float* f1w   = (const float*)d_in[7];
    const float* f1b   = (const float*)d_in[8];
    float* out = (float*)d_out;

    float* ws     = (float*)d_ws;
    float* wbar   = ws;                    // 16384 floats
    float* pooled = ws + K_DIM;            // 512 floats
    float* gates  = ws + K_DIM + M_DIM;    // 512 floats

    wbar_kernel<<<(K_DIM + 255) / 256, 256, 0, stream>>>(c1w, wbar);
    pooled_kernel<<<M_DIM, 256, 0, stream>>>(input, wbar, c1b, pooled);
    gates_kernel<<<1, 512, 0, stream>>>(pooled, f1w, f1b, aw, gates);

    const size_t fp_bytes = (size_t)(K_DIM + 2 * M_DIM) * sizeof(float);       // 69632
    const size_t x_elems  = (size_t)M_DIM * K_DIM;                             // 8.4M
    const size_t w_elems  = (size_t)N_DIM * K_DIM;                             // 41.9M
    const size_t need     = fp_bytes + (x_elems + w_elems) * 2;                // ~96 MB

    dim3 grid(N_DIM / 64, M_DIM / 64);     // 40 x 8 blocks, 4 waves each

    if (ws_size >= need) {
        __bf16* xb  = (__bf16*)((char*)d_ws + fp_bytes);
        __bf16* gwb = xb + x_elems;
        const int x8 = (int)(x_elems / 8);
        const int w8 = (int)(w_elems / 8);
        cvt_bf16_kernel<<<(x8 + 255) / 256, 256, 0, stream>>>(input, xb, x8);
        cvt_bf16_kernel<<<(w8 + 255) / 256, 256, 0, stream>>>(g3w, gwb, w8);
        gemm_bf16_kernel<<<grid, 128, 0, stream>>>(xb, gwb, g3b, gates, out);
    } else {
        gemm_fused_kernel<<<grid, 128, 0, stream>>>(input, g3w, g3b, gates, out);
    }
}
